// ActorNetwork_57286273794162
// MI455X (gfx1250) — matrix operand order repaired and checked by reading, not compiled, not run on hardware
//
#include <hip/hip_runtime.h>

// ---------------------------------------------------------------------------
// ActorNetwork on MI455X (gfx1250, wave32).
//  * Dense GEMMs: v_wmma_f32_16x16x32_bf16. One wave computes a 16x64 C tile
//    (4 accumulators): per 32-wide k-step the wave issues 2x b128 (A frag,
//    reused 4x) + 8x b128 (four pre-packed B frags) + 4 WMMAs, zero VALU.
//    f32 accum seeded with bias, (templated) activation fused in the epilogue.
//  * Sparse GCN aggregation: global f32 atomics (bandwidth-bound,
//    ~0.8 GB @ 23.3 TB/s; dominates the whole net).
// ---------------------------------------------------------------------------

#define E_DIM 128
#define B_GR  256
#define NP_G  500
#define NV_G  50

typedef __attribute__((ext_vector_type(16))) __bf16 v16bf;
typedef __attribute__((ext_vector_type(8)))  float  v8f;

// round-to-nearest-even f32 -> bf16, packed pair (low K in low half, per ISA)
__device__ __forceinline__ unsigned int pack2_bf16(float lo, float hi) {
  unsigned int a = __float_as_uint(lo);
  unsigned int b = __float_as_uint(hi);
  a = (a + 0x7FFFu + ((a >> 16) & 1u)) >> 16;
  b = (b + 0x7FFFu + ((b >> 16) & 1u)) >> 16;
  return (a & 0xFFFFu) | (b << 16);
}

union Frag { unsigned int u[8]; v16bf v; };

// ---------------------------------------------------------------------------
// Weight pre-pack: W[K,N] f32 (rows k >= Korig read as 0) -> bf16 fragments.
// Output dword index: ((k0*tn + tile_n)*32 + lane)*8 + v  with
//   k0 = k>>5, lane = (n&15) | hi<<4, hi = (k>>4)&1, v = (k&15)>>1.
// ---------------------------------------------------------------------------
__global__ void pack_weight_bf16(const float* __restrict__ W, unsigned int* __restrict__ WP,
                                 int K, int N, int Korig) {
  int t = blockIdx.x * blockDim.x + threadIdx.x;
  int total = (K >> 1) * N;
  if (t >= total) return;
  int kp = t / N;                 // K-pair index
  int n  = t - kp * N;
  int k  = kp << 1;
  float x0 = (k     < Korig) ? W[(size_t)k       * N + n] : 0.0f;
  float x1 = (k + 1 < Korig) ? W[(size_t)(k + 1) * N + n] : 0.0f;
  int tn     = N >> 4;
  int tile_n = n >> 4;
  int lane   = (n & 15) | (((k >> 4) & 1) << 4);
  int v      = (k & 15) >> 1;
  size_t idx = ((((size_t)(k >> 5) * tn + tile_n) * 32) + lane) * 8 + v;
  WP[idx] = pack2_bf16(x0, x1);
}

// f32[2*ndw] -> packed bf16 dwords
__global__ void cvt_bf16(const float* __restrict__ x, unsigned int* __restrict__ y, int ndw) {
  int t = blockIdx.x * blockDim.x + threadIdx.x;
  if (t >= ndw) return;
  float2 v = ((const float2*)x)[t];
  y[t] = pack2_bf16(v.x, v.y);
}

// relu fused into the bf16 conversion (f32 copy never needed afterwards)
__global__ void relu_cvt_bf16(const float* __restrict__ x, unsigned int* __restrict__ y, int ndw) {
  int t = blockIdx.x * blockDim.x + threadIdx.x;
  if (t >= ndw) return;
  float2 v = ((const float2*)x)[t];
  y[t] = pack2_bf16(fmaxf(v.x, 0.0f), fmaxf(v.y, 0.0f));
}

// x[M,16] f32 -> y[M,32] bf16 packed (16 dwords/row, upper 8 dwords zero)
__global__ void pad16_bf16(const float* __restrict__ x, unsigned int* __restrict__ y, int M) {
  int t = blockIdx.x * blockDim.x + threadIdx.x;
  if (t >= M * 16) return;
  int r = t >> 4, d = t & 15;
  unsigned int val = 0;
  if (d < 8) {
    float2 v = ((const float2*)x)[r * 8 + d];
    val = pack2_bf16(v.x, v.y);
  }
  y[t] = val;
}

// ---------------------------------------------------------------------------
// C[M,N] = act(Abf[M,K]@W[K,N] + bias); A bf16-packed row-major, W pre-packed
// fragments. K % 32 == 0, N % 64 == 0. One wave per 16x64 C tile (A frag
// reused across 4 WMMAs). ACT: 0=none, 2=leaky(0.01).
// ---------------------------------------------------------------------------
template <int ACT>
__global__ void __launch_bounds__(256)
gemm_bf16_wmma(const unsigned int* __restrict__ Abf, const unsigned int* __restrict__ BP,
               const float* __restrict__ bias, float* __restrict__ C,
               int M, int N, int K) {
  const int lane = threadIdx.x & 31;
  const int wid  = (blockIdx.x * blockDim.x + threadIdx.x) >> 5;
  const int tn4  = N >> 6;                    // 64-col tiles per row
  const int tile_m  = wid / tn4;
  const int tile_n4 = wid - tile_m * tn4;
  if (tile_m >= (M >> 4)) return;             // whole-wave exit: EXEC all-1s
  const int m0 = tile_m << 4, n0 = tile_n4 << 6;
  const int l  = lane & 15;                   // A: M row / B,C: N col
  const int hi = lane >> 4;                   // half-wave select

  v8f acc[4];
#pragma unroll
  for (int j = 0; j < 4; ++j) {
    const float bj = bias ? bias[n0 + (j << 4) + l] : 0.0f;
#pragma unroll
    for (int r = 0; r < 8; ++r) acc[j][r] = bj;
  }

  const uint4* __restrict__ Arow = (const uint4*)(Abf + (size_t)(m0 + l) * (K >> 1));
  const int tn = N >> 4;
  // B frag for (ks, tile_n4*4+j): uint4 index ((ks*tn + tile_n4*4 + j)*32 + lane)*2
  const uint4* __restrict__ Bfr =
      (const uint4*)BP + ((((size_t)tile_n4 * 4) * 32 + lane) << 1);
  const size_t bStep = (size_t)tn * 64;       // uint4s per k-step over all tiles

  const int kSteps = K >> 5;
  for (int ks = 0; ks < kSteps; ++ks) {
    Frag af;
    // A frag dwords v=0..7 -> K-pairs ks*16 + hi*8 + v
    uint4 a0 = Arow[(ks << 2) + (hi << 1)];
    uint4 a1 = Arow[(ks << 2) + (hi << 1) + 1];
    af.u[0] = a0.x; af.u[1] = a0.y; af.u[2] = a0.z; af.u[3] = a0.w;
    af.u[4] = a1.x; af.u[5] = a1.y; af.u[6] = a1.z; af.u[7] = a1.w;
#pragma unroll
    for (int j = 0; j < 4; ++j) {
      Frag bf;
      uint4 b0 = Bfr[(size_t)j * 64];
      uint4 b1 = Bfr[(size_t)j * 64 + 1];
      bf.u[0] = b0.x; bf.u[1] = b0.y; bf.u[2] = b0.z; bf.u[3] = b0.w;
      bf.u[4] = b1.x; bf.u[5] = b1.y; bf.u[6] = b1.z; bf.u[7] = b1.w;
      acc[j] = __builtin_amdgcn_wmma_f32_16x16x32_bf16(false, af.v, false, bf.v,
                                                       (short)0, acc[j], false, false);
    }
    Bfr += bStep;
  }
  // C f32: lane -> n = n0 + j*16 + l ; VGPR r -> m = m0 + r + 8*hi
#pragma unroll
  for (int j = 0; j < 4; ++j) {
#pragma unroll
    for (int r = 0; r < 8; ++r) {
      float vv = acc[j][r];
      if (ACT == 2) vv = vv > 0.0f ? vv : 0.01f * vv;
      C[(size_t)(m0 + (hi << 3) + r) * N + n0 + (j << 4) + l] = vv;
    }
  }
}

// ------------------------- sparse / elementwise helpers --------------------

__global__ void fill_value(float* __restrict__ p, float v, int n) {
  int t = blockIdx.x * blockDim.x + threadIdx.x;
  if (t < n) p[t] = v;
}

__global__ void deg_scatter(const int* __restrict__ dst, float* __restrict__ deg, int nE) {
  int t = blockIdx.x * blockDim.x + threadIdx.x;
  if (t < nE) atomicAdd(deg + dst[t], 1.0f);
}

__global__ void rsqrt_inplace(float* __restrict__ p, int n) {
  int t = blockIdx.x * blockDim.x + threadIdx.x;
  if (t < n) p[t] = rsqrtf(p[t]);
}

// acc[i,f] = xw[i,f] * dinv[i]^2 + bias[f]   (self-loop term + bias)
__global__ void gcn_self_init(const float* __restrict__ xw, const float* __restrict__ dinv,
                              const float* __restrict__ bias, float* __restrict__ acc, int total) {
  int t = blockIdx.x * blockDim.x + threadIdx.x;
  if (t >= total) return;
  int i = t >> 7, f = t & (E_DIM - 1);
  float di = dinv[i];
  acc[t] = xw[t] * di * di + bias[f];
}

// acc[dst] += xw[src] * dinv[src]*dinv[dst] ; 32 lanes/edge, float4 chunks
__global__ void gcn_edge_scatter(const float* __restrict__ xw, const int* __restrict__ src,
                                 const int* __restrict__ dst, const float* __restrict__ dinv,
                                 float* __restrict__ acc, int nE) {
  int t = blockIdx.x * blockDim.x + threadIdx.x;
  int e = t >> 5;
  if (e >= nE) return;
  int f4 = (t & 31) << 2;
  int s = src[e], d = dst[e];
  float c = dinv[s] * dinv[d];
  const float4 x = *(const float4*)(xw + (size_t)s * E_DIM + f4);
  float* o = acc + (size_t)d * E_DIM + f4;
  atomicAdd(o + 0, x.x * c);
  atomicAdd(o + 1, x.y * c);
  atomicAdd(o + 2, x.z * c);
  atomicAdd(o + 3, x.w * c);
}

// graph[b,f] = mean over npg nodes; one block per graph, 128 threads (=E)
__global__ void graph_mean(const float* __restrict__ h, float* __restrict__ g, int npg) {
  int b = blockIdx.x, f = threadIdx.x;
  const float* base = h + (size_t)b * npg * E_DIM + f;
  float s = 0.0f;
  for (int n = 0; n < npg; ++n) s += base[(size_t)n * E_DIM];
  g[b * E_DIM + f] = s * (1.0f / (float)npg);
}

// head input [VN,512] -> bf16 packed dwords directly:
// row r: [v_h2[r] | v_h0[r] | v_graph[b] | p_graph[b]], b = r / NV_G
__global__ void build_head_in_bf16(const float* __restrict__ vh2, const float* __restrict__ vh0,
                                   const float* __restrict__ vg, const float* __restrict__ pg,
                                   unsigned int* __restrict__ x4, int rows) {
  int t = blockIdx.x * blockDim.x + threadIdx.x;
  if (t >= rows * 256) return;          // one packed dword = 2 consecutive cols
  int r = t >> 8, cp = t & 255;
  int c = cp << 1;                      // pairs never straddle the 128-col segments
  int b = r / NV_G;
  const float* s;
  int o;
  if      (c < 128) { s = vh2 + (size_t)r * E_DIM; o = c; }
  else if (c < 256) { s = vh0 + (size_t)r * E_DIM; o = c - 128; }
  else if (c < 384) { s = vg + (size_t)b * E_DIM;  o = c - 256; }
  else              { s = pg + (size_t)b * E_DIM;  o = c - 384; }
  x4[t] = pack2_bf16(s[o], s[o + 1]);
}

// logits[r] = dot(h[r,:], w3) + b3 ; one block (128 threads) per row
__global__ void head_out(const float* __restrict__ h, const float* __restrict__ w3,
                         const float* __restrict__ b3, float* __restrict__ out) {
  __shared__ float red[E_DIM];
  int r = blockIdx.x, f = threadIdx.x;
  red[f] = h[(size_t)r * E_DIM + f] * w3[f];
  __syncthreads();
  for (int s = 64; s > 0; s >>= 1) {
    if (f < s) red[f] += red[f + s];
    __syncthreads();
  }
  if (f == 0) out[r] = red[0] + b3[0];
}

// ------------------------------- launcher ----------------------------------

static inline void launch_gemm(const unsigned int* Ab, const unsigned int* BP,
                               const float* bias, float* C, int M, int N, int K,
                               int act, hipStream_t s) {
  int tiles  = (M >> 4) * (N >> 6);      // 16x64 tiles
  int blocks = (tiles + 7) / 8;          // 8 waves (256 thr) per block
  if (act == 2) gemm_bf16_wmma<2><<<blocks, 256, 0, s>>>(Ab, BP, bias, C, M, N, K);
  else          gemm_bf16_wmma<0><<<blocks, 256, 0, s>>>(Ab, BP, bias, C, M, N, K);
}

extern "C" void kernel_launch(void* const* d_in, const int* in_sizes, int n_in,
                              void* d_out, int out_size, void* d_ws, size_t ws_size,
                              hipStream_t stream) {
  (void)in_sizes; (void)n_in; (void)out_size; (void)ws_size;

  const float* p_x  = (const float*)d_in[0];
  const float* v_x  = (const float*)d_in[1];
  const int*   p_ei = (const int*)d_in[2];
  const int*   v_ei = (const int*)d_in[3];
  // d_in[4], d_in[5]: batch ids (graphs are equal-sized; implicit)
  const float* pW0 = (const float*)d_in[6];  const float* pb0 = (const float*)d_in[7];
  const float* pW1 = (const float*)d_in[8];  const float* pb1 = (const float*)d_in[9];
  const float* pW2 = (const float*)d_in[10]; const float* pb2 = (const float*)d_in[11];
  const float* vW0 = (const float*)d_in[12]; const float* vb0 = (const float*)d_in[13];
  const float* vW1 = (const float*)d_in[14]; const float* vb1 = (const float*)d_in[15];
  const float* vW2 = (const float*)d_in[16]; const float* vb2 = (const float*)d_in[17];
  const float* hW1 = (const float*)d_in[18]; const float* hb1 = (const float*)d_in[19];
  const float* hW2 = (const float*)d_in[20]; const float* hb2 = (const float*)d_in[21];
  const float* hW3 = (const float*)d_in[22]; const float* hb3 = (const float*)d_in[23];

  const int PN  = B_GR * NP_G;            // 128000 particle nodes
  const int VN  = B_GR * NV_G;            // 12800 vehicle nodes
  const int PEe = B_GR * 6 * NP_G;        // 768000 p-edges
  const int VEe = B_GR * 4 * NV_G;        // 51200 v-edges
  const int PNE = PN * E_DIM;
  const int VNE = VN * E_DIM;

  const int* p_src = p_ei;        const int* p_dst = p_ei + PEe;
  const int* v_src = v_ei;        const int* v_dst = v_ei + VEe;

  // ---- workspace carve (256B aligned) ----
  char* wsp = (char*)d_ws;
  auto takeB = [&](size_t bytes) {
    void* p = wsp;
    wsp += (bytes + 255) & ~(size_t)255;
    return p;
  };
  auto take  = [&](size_t nf) { return (float*)takeB(nf * sizeof(float)); };
  auto takeU = [&](size_t nu) { return (unsigned int*)takeB(nu * sizeof(unsigned int)); };

  float* p_dinv = take(PN);
  float* pA     = take(PNE);              // rotating f32 node-feature buffers
  float* pB     = take(PNE);
  float* pC     = take(PNE);
  float* v_dinv = take(VN);
  float* v_h0   = take(VNE);              // kept for head concat
  float* vA     = take(VNE);
  float* vB     = take(VNE);
  float* vC     = take(VNE);              // h2, kept for head concat
  float* p_grph = take(B_GR * E_DIM);
  float* v_grph = take(B_GR * E_DIM);
  float* hH1    = take((size_t)VN * 256);
  float* hH2    = take(VNE);
  // bf16-packed GEMM A operands (dwords = M*K/2)
  unsigned int* pXb  = takeU((size_t)PN * 16);   // padded K=32
  unsigned int* pGb  = takeU((size_t)PN * 64);   // h0 / h1 bf16 (reused)
  unsigned int* vXb  = takeU((size_t)VN * 16);
  unsigned int* vGb  = takeU((size_t)VN * 64);
  unsigned int* x4b  = takeU((size_t)VN * 256);
  unsigned int* hH1b = takeU((size_t)VN * 128);
  // packed bf16 weights (dwords = K*N/2, K padded to mult of 32)
  unsigned int* pW0p = takeU(32 * 128 / 2);
  unsigned int* pW1p = takeU(128 * 128 / 2);
  unsigned int* pW2p = takeU(128 * 128 / 2);
  unsigned int* vW0p = takeU(32 * 128 / 2);
  unsigned int* vW1p = takeU(128 * 128 / 2);
  unsigned int* vW2p = takeU(128 * 128 / 2);
  unsigned int* hW1p = takeU(512 * 256 / 2);
  unsigned int* hW2p = takeU(256 * 128 / 2);

  auto cdiv = [](int a, int b) { return (a + b - 1) / b; };
  auto packW = [&](const float* W, unsigned int* WP, int K, int N, int Ko) {
    pack_weight_bf16<<<cdiv((K >> 1) * N, 256), 256, 0, stream>>>(W, WP, K, N, Ko);
  };

  // ---------------- weight packing + input padding -------------------------
  packW(pW0, pW0p, 32, 128, 16);   packW(pW1, pW1p, 128, 128, 128);
  packW(pW2, pW2p, 128, 128, 128); packW(vW0, vW0p, 32, 128, 16);
  packW(vW1, vW1p, 128, 128, 128); packW(vW2, vW2p, 128, 128, 128);
  packW(hW1, hW1p, 512, 256, 512); packW(hW2, hW2p, 256, 128, 256);
  pad16_bf16<<<cdiv(PN * 16, 256), 256, 0, stream>>>(p_x, pXb, PN);
  pad16_bf16<<<cdiv(VN * 16, 256), 256, 0, stream>>>(v_x, vXb, VN);

  // ---------------- particle encoder (only p_graph is consumed) -----------
  fill_value<<<cdiv(PN, 256), 256, 0, stream>>>(p_dinv, 1.0f, PN);       // self loop
  deg_scatter<<<cdiv(PEe, 256), 256, 0, stream>>>(p_dst, p_dinv, PEe);
  rsqrt_inplace<<<cdiv(PN, 256), 256, 0, stream>>>(p_dinv, PN);

  launch_gemm(pXb, pW0p, pb0, pA, PN, E_DIM, 32, 0, stream);             // h0
  cvt_bf16<<<cdiv(PNE / 2, 256), 256, 0, stream>>>(pA, pGb, PNE / 2);
  launch_gemm(pGb, pW1p, nullptr, pB, PN, E_DIM, E_DIM, 0, stream);      // xw1
  gcn_self_init<<<cdiv(PNE, 256), 256, 0, stream>>>(pB, p_dinv, pb1, pC, PNE);
  gcn_edge_scatter<<<cdiv(PEe * 32, 256), 256, 0, stream>>>(pB, p_src, p_dst, p_dinv, pC, PEe);
  relu_cvt_bf16<<<cdiv(PNE / 2, 256), 256, 0, stream>>>(pC, pGb, PNE / 2);  // h1 (bf16 only)
  launch_gemm(pGb, pW2p, nullptr, pA, PN, E_DIM, E_DIM, 0, stream);      // xw2
  gcn_self_init<<<cdiv(PNE, 256), 256, 0, stream>>>(pA, p_dinv, pb2, pB, PNE);
  gcn_edge_scatter<<<cdiv(PEe * 32, 256), 256, 0, stream>>>(pA, p_src, p_dst, p_dinv, pB, PEe);
  graph_mean<<<B_GR, E_DIM, 0, stream>>>(pB, p_grph, NP_G);

  // ---------------- vehicle encoder (needs h0, h2, graph) -----------------
  fill_value<<<cdiv(VN, 256), 256, 0, stream>>>(v_dinv, 1.0f, VN);
  deg_scatter<<<cdiv(VEe, 256), 256, 0, stream>>>(v_dst, v_dinv, VEe);
  rsqrt_inplace<<<cdiv(VN, 256), 256, 0, stream>>>(v_dinv, VN);

  launch_gemm(vXb, vW0p, vb0, v_h0, VN, E_DIM, 32, 0, stream);           // h0 (kept)
  cvt_bf16<<<cdiv(VNE / 2, 256), 256, 0, stream>>>(v_h0, vGb, VNE / 2);
  launch_gemm(vGb, vW1p, nullptr, vA, VN, E_DIM, E_DIM, 0, stream);      // xw1
  gcn_self_init<<<cdiv(VNE, 256), 256, 0, stream>>>(vA, v_dinv, vb1, vB, VNE);
  gcn_edge_scatter<<<cdiv(VEe * 32, 256), 256, 0, stream>>>(vA, v_src, v_dst, v_dinv, vB, VEe);
  relu_cvt_bf16<<<cdiv(VNE / 2, 256), 256, 0, stream>>>(vB, vGb, VNE / 2);  // h1 (bf16 only)
  launch_gemm(vGb, vW2p, nullptr, vA, VN, E_DIM, E_DIM, 0, stream);      // xw2
  gcn_self_init<<<cdiv(VNE, 256), 256, 0, stream>>>(vA, v_dinv, vb2, vC, VNE);
  gcn_edge_scatter<<<cdiv(VEe * 32, 256), 256, 0, stream>>>(vA, v_src, v_dst, v_dinv, vC, VEe);
  graph_mean<<<B_GR, E_DIM, 0, stream>>>(vC, v_grph, NV_G);

  // ---------------- MLP head ----------------------------------------------
  build_head_in_bf16<<<cdiv(VN * 256, 256), 256, 0, stream>>>(vC, v_h0, v_grph, p_grph, x4b, VN);
  launch_gemm(x4b, hW1p, hb1, hH1, VN, 256, 512, 2, stream);             // leaky_relu
  cvt_bf16<<<cdiv(VN * 128, 256), 256, 0, stream>>>(hH1, hH1b, VN * 128);
  launch_gemm(hH1b, hW2p, hb2, hH2, VN, E_DIM, 256, 2, stream);          // leaky_relu
  head_out<<<VN, E_DIM, 0, stream>>>(hH2, hW3, hb3, (float*)d_out);      // [B, NV]
}
